// SpMiddleResNetFHD_3856880632252
// MI455X (gfx1250) — matrix operand order repaired and checked by reading, not compiled, not run on hardware
//
#include <hip/hip_runtime.h>
#include <hip/hip_bf16.h>

typedef __attribute__((ext_vector_type(16))) _Float16 v16h;
typedef __attribute__((ext_vector_type(8)))  _Float16 v8h;
typedef __attribute__((ext_vector_type(8)))  float    v8f;

static inline size_t alignup(size_t x) { return (x + 255) & ~(size_t)255; }

// ---------------- utility kernels ----------------

__global__ void k_bn_fold(const float* __restrict__ g, const float* __restrict__ b,
                          const float* __restrict__ m, const float* __restrict__ v,
                          float* __restrict__ scale, float* __restrict__ shift, int c) {
  int i = blockIdx.x * blockDim.x + threadIdx.x;
  if (i < c) {
    float sc = g[i] * rsqrtf(v[i] + 1e-3f);
    scale[i] = sc;
    shift[i] = b[i] - m[i] * sc;
  }
}

__global__ void k_scatter(const float* __restrict__ vf, const int* __restrict__ co,
                          _Float16* __restrict__ x0, _Float16* __restrict__ m0,
                          int N, int D, int H, int W, int C) {
  int i = blockIdx.x * blockDim.x + threadIdx.x;
  if (i >= N) return;
  int z = co[i * 4 + 1], y = co[i * 4 + 2], x = co[i * 4 + 3];
  int pos = (z * H + y) * W + x;
  for (int c = 0; c < C; ++c) x0[pos * C + c] = (_Float16)vf[i * C + c];
  m0[pos] = (_Float16)1.0f;
}

__global__ void k_final(const _Float16* __restrict__ a, float* __restrict__ o,
                        int D, int H, int W, int C) {
  int t = blockIdx.x * blockDim.x + threadIdx.x;
  int tot = C * D * H * W;
  if (t >= tot) return;
  int w = t % W, h = (t / W) % H, d = (t / (W * H)) % D, c = t / (W * H * D);
  o[t] = (float)a[((d * H + h) * W + w) * C + c];
}

// Pack conv weights (f32, layout (taps, CIN, Cout)) into per-wave WMMA B-fragment
// order (f16): p[(((nt*NCH + ch)*32 + lane)*16) + j] = w[vk*Cout + co], where
// co = nt*16 + (lane&15), vk = ch*32 + (lane>>4)*8 + (j<8 ? j : j+8). Each conv lane then reads
// its whole B fragment as one contiguous 32-byte load.
__global__ void k_pack_w(const float* __restrict__ w, _Float16* __restrict__ p,
                         int Cout, int NCH, int KTOT) {
  int i = blockIdx.x * blockDim.x + threadIdx.x;
  int tot = (Cout / 16) * NCH * 512;
  if (i >= tot) return;
  int j = i & 15;
  int lane = (i >> 4) & 31;
  int rest = i >> 9;
  int ch = rest % NCH;
  int nt = rest / NCH;
  int co = nt * 16 + (lane & 15);
  int vk = ch * 32 + (lane >> 4) * 8 + (j < 8 ? j : j + 8);
  p[i] = (vk < KTOT) ? (_Float16)w[vk * Cout + co] : (_Float16)0.f;
}

// ---------------- implicit-GEMM WMMA conv ----------------
// One wave = 16(position) x 16(out-channel) tile; NW waves/block share the same
// 16 positions and cover Cout = NW*16 channels.
// NW==1 : direct global A gathers (two predicated global_load_b128 per chunk).
// NW>1  : waves cooperatively gather an 8-chunk window of A fragments into LDS
//         (8 KB), sync, then every wave consumes the window via ds_load_b128 +
//         v_wmma — removing the NWx redundant global A traffic.
// B fragment: one contiguous 32B load/lane from the packed weight stream.

template <int CIN, int KD, int KH, int KW, int LW, int LH, int NW, bool DOWN, bool RESID>
__global__ void __launch_bounds__(256)
k_conv_wmma(const _Float16* __restrict__ xin,
            const _Float16* __restrict__ mask_in,
            const _Float16* __restrict__ wpack,
            const float* __restrict__ bnscale,
            const float* __restrict__ bnshift,
            const _Float16* __restrict__ resid,     // used iff RESID
            _Float16* __restrict__ yout,            // (Mout, Cout)
            _Float16* __restrict__ mask_out,        // used iff DOWN
            int inD, int inH, int inW,
            int outD,
            int sd, int sh, int sw,
            int pd, int ph, int pw) {
  constexpr int KTAPS = KD * KH * KW;
  constexpr int KTOT = KTAPS * CIN;
  constexpr int NCH = (KTOT + 31) / 32;
  constexpr int Cout = NW * 16;
  constexpr int outW = 1 << LW;
  constexpr int outH = 1 << LH;
  constexpr int LWH = LW + LH;
  constexpr int WIN = 8;                       // chunks per LDS window (NW | WIN)

  __shared__ _Float16 sA[NW > 1 ? WIN * 512 : 1];

  const int lane = threadIdx.x & 31;
  const int wv = threadIdx.x >> 5;
  const int co = wv * 16 + (lane & 15);
  const int Mout = outD << LWH;
  const int mA = blockIdx.x * 16 + (lane & 15);
  const int kgrpA = (lane >> 4) * 8;

  const bool mAok = (mA < Mout);
  int ibz = 0, iby = 0, ibx = 0;
  if (mAok) {
    const int ox = mA & (outW - 1);
    const int oy = (mA >> LW) & (outH - 1);
    const int oz = mA >> LWH;
    ibz = oz * sd - pd;
    iby = oy * sh - ph;
    ibx = ox * sw - pw;
  }

  // Gather one 32-K chunk of the A fragment for this lane.
  auto gatherA = [&](int ch) -> v16h {
    v16h a;
    if constexpr (CIN % 8 == 0) {
#pragma unroll
      for (int g = 0; g < 2; ++g) {
        const int vk0 = ch * 32 + kgrpA + g * 16;  // 8 consecutive K, one tap
        const int tap = vk0 / CIN;
        const int ci = vk0 - tap * CIN;
        v8h av = {};
        if (tap < KTAPS && mAok) {
          const int kd = tap / (KH * KW);
          const int r2 = tap - kd * (KH * KW);
          const int kh = r2 / KW;
          const int kw = r2 - kh * KW;
          const int iz = ibz + kd, iy = iby + kh, ix = ibx + kw;
          if ((unsigned)iz < (unsigned)inD && (unsigned)iy < (unsigned)inH &&
              (unsigned)ix < (unsigned)inW)
            av = *(const v8h*)(xin + ((iz * inH + iy) * inW + ix) * CIN + ci);
        }
#pragma unroll
        for (int j = 0; j < 8; ++j) a[g * 8 + j] = av[j];
      }
    } else {
#pragma unroll
      for (int j = 0; j < 16; ++j) {
        const int vk = ch * 32 + kgrpA + (j < 8 ? j : 8 + j);
        const int tap = vk / CIN;
        const int ci = vk - tap * CIN;
        _Float16 av = (_Float16)0.f;
        if (tap < KTAPS && mAok) {
          const int kd = tap / (KH * KW);
          const int r2 = tap - kd * (KH * KW);
          const int kh = r2 / KW;
          const int kw = r2 - kh * KW;
          const int iz = ibz + kd, iy = iby + kh, ix = ibx + kw;
          if ((unsigned)iz < (unsigned)inD && (unsigned)iy < (unsigned)inH &&
              (unsigned)ix < (unsigned)inW)
            av = xin[((iz * inH + iy) * inW + ix) * CIN + ci];
        }
        a[j] = av;
      }
    }
    return a;
  };

  const _Float16* bptr = wpack + ((size_t)(wv * NCH) * 32 + lane) * 16;

  v8f acc = {};
  if constexpr (NW > 1) {
    for (int win = 0; win < NCH; win += WIN) {
      // producer phase: each wave fills its slots of the window
      for (int s = wv; s < WIN; s += NW) {
        const int ch = win + s;
        if (ch < NCH) {
          v16h a = gatherA(ch);
          *(v16h*)(&sA[s * 512 + lane * 16]) = a;
        }
      }
      __syncthreads();
      // consumer phase: every wave runs all chunks of the window
#pragma unroll
      for (int s = 0; s < WIN; ++s) {
        const int ch = win + s;
        if (ch < NCH) {
          v16h a = *(const v16h*)(&sA[s * 512 + lane * 16]);
          v16h b = *(const v16h*)(bptr + (size_t)ch * 512);
          if (ch + 1 < NCH)
            __builtin_prefetch(bptr + (size_t)(ch + 1) * 512, 0, 3);
          acc = __builtin_amdgcn_wmma_f32_16x16x32_f16(false, a, false, b,
                                                       (short)0, acc, false, false);
        }
      }
      __syncthreads();
    }
  } else {
    for (int ch = 0; ch < NCH; ++ch) {
      v16h b = *(const v16h*)(bptr + (size_t)ch * 512);
      if (ch + 1 < NCH)
        __builtin_prefetch(bptr + (size_t)(ch + 1) * 512, 0, 3);
      v16h a = gatherA(ch);
      acc = __builtin_amdgcn_wmma_f32_16x16x32_f16(false, a, false, b,
                                                   (short)0, acc, false, false);
    }
  }

  const float sc = bnscale[co];
  const float sf = bnshift[co];
  const int mrow = blockIdx.x * 16 + 8 * (lane >> 4);

#pragma unroll
  for (int v = 0; v < 8; ++v) {
    const int m = mrow + v;
    if (m >= Mout) continue;
    float r = acc[v] * sc + sf;
    float mv;
    if (DOWN) {
      const int x = m & (outW - 1);
      const int y = (m >> LW) & (outH - 1);
      const int z = m >> LWH;
      const int bz = z * sd - pd, by = y * sh - ph, bx = x * sw - pw;
      float occ = 0.f;
#pragma unroll
      for (int kd = 0; kd < KD; ++kd)
#pragma unroll
        for (int kh = 0; kh < KH; ++kh)
#pragma unroll
          for (int kw = 0; kw < KW; ++kw) {
            const int iz = bz + kd, iy = by + kh, ix = bx + kw;
            if ((unsigned)iz < (unsigned)inD && (unsigned)iy < (unsigned)inH &&
                (unsigned)ix < (unsigned)inW)
              occ += (float)mask_in[(iz * inH + iy) * inW + ix];
          }
      mv = occ > 0.f ? 1.f : 0.f;
      if (wv == 0 && (lane & 15) == 0) mask_out[m] = (_Float16)mv;
    } else {
      mv = (float)mask_in[m];
    }
    r *= mv;
    if (RESID) r += (float)resid[m * Cout + co];
    r = r > 0.f ? r : 0.f;
    yout[m * Cout + co] = (_Float16)r;
  }
}

// ---------------- host orchestration ----------------

extern "C" void kernel_launch(void* const* d_in, const int* in_sizes, int n_in,
                              void* d_out, int out_size, void* d_ws, size_t ws_size,
                              hipStream_t stream) {
  (void)in_sizes; (void)n_in; (void)out_size; (void)ws_size;

  constexpr int D0 = 41, H0 = 256, W0 = 256; constexpr int M0 = D0 * H0 * W0;
  constexpr int D1 = 21, H1 = 128, W1 = 128; constexpr int M1 = D1 * H1 * W1;
  constexpr int D2 = 11, H2 = 64, W2 = 64;   constexpr int M2 = D2 * H2 * W2;
  constexpr int D3 = 5,  H3 = 32, W3 = 32;   constexpr int M3 = D3 * H3 * W3;
  constexpr int D4 = 2,  H4 = 32, W4 = 32;   constexpr int M4 = D4 * H4 * W4;
  constexpr int NVOX = 60000;

  // K totals and chunk counts per conv flavor
  constexpr int KT_C0 = 27 * 5,    NC_C0 = (KT_C0 + 31) / 32;     // 135 -> 5
  constexpr int KT_16 = 27 * 16,   NC_16 = (KT_16 + 31) / 32;     // 432 -> 14
  constexpr int KT_32 = 27 * 32,   NC_32 = (KT_32 + 31) / 32;     // 864 -> 27
  constexpr int KT_64 = 27 * 64,   NC_64 = (KT_64 + 31) / 32;     // 1728 -> 54
  constexpr int KT_128 = 27 * 128, NC_128 = (KT_128 + 31) / 32;   // 3456 -> 108
  constexpr int KT_D4 = 3 * 128,   NC_D4 = (KT_D4 + 31) / 32;     // 384 -> 12

  // Packed-weight element counts: (Cout/16) * NCH * 512
  constexpr int P_C0 = 1 * NC_C0 * 512;
  constexpr int P_R16 = 1 * NC_16 * 512;
  constexpr int P_D1 = 2 * NC_16 * 512;
  constexpr int P_R32 = 2 * NC_32 * 512;
  constexpr int P_D2 = 4 * NC_32 * 512;
  constexpr int P_R64 = 4 * NC_64 * 512;
  constexpr int P_D3 = 8 * NC_64 * 512;
  constexpr int P_R128 = 8 * NC_128 * 512;
  constexpr int P_D4 = 8 * NC_D4 * 512;

  char* ws = (char*)d_ws;
  size_t off = 0;
  auto bump = [&](size_t bytes) -> char* { char* p = ws + off; off = alignup(off + bytes); return p; };

  _Float16* x5  = (_Float16*)bump((size_t)M0 * 5 * 2);
  _Float16* m0  = (_Float16*)bump((size_t)M0 * 2);
  _Float16* A16 = (_Float16*)bump((size_t)M0 * 16 * 2);

  _Float16* wC0 = (_Float16*)bump(P_C0 * 2);
  _Float16* wR16[4];  for (int i = 0; i < 4; ++i) wR16[i]  = (_Float16*)bump(P_R16 * 2);
  _Float16* wD1 = (_Float16*)bump(P_D1 * 2);
  _Float16* wR32[4];  for (int i = 0; i < 4; ++i) wR32[i]  = (_Float16*)bump(P_R32 * 2);
  _Float16* wD2 = (_Float16*)bump(P_D2 * 2);
  _Float16* wR64[4];  for (int i = 0; i < 4; ++i) wR64[i]  = (_Float16*)bump(P_R64 * 2);
  _Float16* wD3 = (_Float16*)bump(P_D3 * 2);
  _Float16* wR128[4]; for (int i = 0; i < 4; ++i) wR128[i] = (_Float16*)bump(P_R128 * 2);
  _Float16* wD4 = (_Float16*)bump(P_D4 * 2);

  float* sc0 = (float*)bump(16 * 4); float* sh0 = (float*)bump(16 * 4);
  float *scR16[4], *shR16[4];
  for (int i = 0; i < 4; ++i) { scR16[i] = (float*)bump(16 * 4); shR16[i] = (float*)bump(16 * 4); }
  float* scD1 = (float*)bump(32 * 4); float* shD1 = (float*)bump(32 * 4);
  float *scR32[4], *shR32[4];
  for (int i = 0; i < 4; ++i) { scR32[i] = (float*)bump(32 * 4); shR32[i] = (float*)bump(32 * 4); }
  float* scD2 = (float*)bump(64 * 4); float* shD2 = (float*)bump(64 * 4);
  float *scR64[4], *shR64[4];
  for (int i = 0; i < 4; ++i) { scR64[i] = (float*)bump(64 * 4); shR64[i] = (float*)bump(64 * 4); }
  float* scD3 = (float*)bump(128 * 4); float* shD3 = (float*)bump(128 * 4);
  float *scR128[4], *shR128[4];
  for (int i = 0; i < 4; ++i) { scR128[i] = (float*)bump(128 * 4); shR128[i] = (float*)bump(128 * 4); }
  float* scD4 = (float*)bump(128 * 4); float* shD4 = (float*)bump(128 * 4);

  _Float16* B16 = (_Float16*)bump((size_t)M0 * 16 * 2);
  _Float16* C16 = (_Float16*)bump((size_t)M0 * 16 * 2);

  // Later-stage buffers alias the B16/C16 region (dead once stage 1 starts).
  size_t off2 = (size_t)((char*)B16 - ws);
  auto bump2 = [&](size_t bytes) -> char* { char* p = ws + off2; off2 = alignup(off2 + bytes); return p; };
  _Float16* m1   = (_Float16*)bump2((size_t)M1 * 2);
  _Float16* A32  = (_Float16*)bump2((size_t)M1 * 32 * 2);
  _Float16* B32  = (_Float16*)bump2((size_t)M1 * 32 * 2);
  _Float16* C32  = (_Float16*)bump2((size_t)M1 * 32 * 2);
  _Float16* m2   = (_Float16*)bump2((size_t)M2 * 2);
  _Float16* A64  = (_Float16*)bump2((size_t)M2 * 64 * 2);
  _Float16* B64  = (_Float16*)bump2((size_t)M2 * 64 * 2);
  _Float16* C64  = (_Float16*)bump2((size_t)M2 * 64 * 2);
  _Float16* m3   = (_Float16*)bump2((size_t)M3 * 2);
  _Float16* A128 = (_Float16*)bump2((size_t)M3 * 128 * 2);
  _Float16* B128 = (_Float16*)bump2((size_t)M3 * 128 * 2);
  _Float16* C128 = (_Float16*)bump2((size_t)M3 * 128 * 2);
  _Float16* m4   = (_Float16*)bump2((size_t)M4 * 2);
  _Float16* F    = (_Float16*)bump2((size_t)M4 * 128 * 2);

  auto pack = [&](int idx, _Float16* dst, int Cout, int NCH, int KTOT) {
    int tot = (Cout / 16) * NCH * 512;
    k_pack_w<<<dim3((tot + 255) / 256), dim3(256), 0, stream>>>(
        (const float*)d_in[idx], dst, Cout, NCH, KTOT);
  };
  auto bnf = [&](int idx, float* sc, float* sf, int c) {
    k_bn_fold<<<dim3(1), dim3(128), 0, stream>>>(
        (const float*)d_in[idx], (const float*)d_in[idx + 1],
        (const float*)d_in[idx + 2], (const float*)d_in[idx + 3], sc, sf, c);
  };

  hipMemsetAsync(x5, 0, (size_t)M0 * 5 * 2, stream);
  hipMemsetAsync(m0, 0, (size_t)M0 * 2, stream);

  pack(1, wC0, 16, NC_C0, KT_C0);
  pack(6, wR16[0], 16, NC_16, KT_16);  pack(11, wR16[1], 16, NC_16, KT_16);
  pack(16, wR16[2], 16, NC_16, KT_16); pack(21, wR16[3], 16, NC_16, KT_16);
  pack(26, wD1, 32, NC_16, KT_16);
  pack(31, wR32[0], 32, NC_32, KT_32); pack(36, wR32[1], 32, NC_32, KT_32);
  pack(41, wR32[2], 32, NC_32, KT_32); pack(46, wR32[3], 32, NC_32, KT_32);
  pack(51, wD2, 64, NC_32, KT_32);
  pack(56, wR64[0], 64, NC_64, KT_64); pack(61, wR64[1], 64, NC_64, KT_64);
  pack(66, wR64[2], 64, NC_64, KT_64); pack(71, wR64[3], 64, NC_64, KT_64);
  pack(76, wD3, 128, NC_64, KT_64);
  pack(81, wR128[0], 128, NC_128, KT_128); pack(86, wR128[1], 128, NC_128, KT_128);
  pack(91, wR128[2], 128, NC_128, KT_128); pack(96, wR128[3], 128, NC_128, KT_128);
  pack(101, wD4, 128, NC_D4, KT_D4);

  bnf(2, sc0, sh0, 16);
  bnf(7, scR16[0], shR16[0], 16);  bnf(12, scR16[1], shR16[1], 16);
  bnf(17, scR16[2], shR16[2], 16); bnf(22, scR16[3], shR16[3], 16);
  bnf(27, scD1, shD1, 32);
  bnf(32, scR32[0], shR32[0], 32); bnf(37, scR32[1], shR32[1], 32);
  bnf(42, scR32[2], shR32[2], 32); bnf(47, scR32[3], shR32[3], 32);
  bnf(52, scD2, shD2, 64);
  bnf(57, scR64[0], shR64[0], 64); bnf(62, scR64[1], shR64[1], 64);
  bnf(67, scR64[2], shR64[2], 64); bnf(72, scR64[3], shR64[3], 64);
  bnf(77, scD3, shD3, 128);
  bnf(82, scR128[0], shR128[0], 128); bnf(87, scR128[1], shR128[1], 128);
  bnf(92, scR128[2], shR128[2], 128); bnf(97, scR128[3], shR128[3], 128);
  bnf(102, scD4, shD4, 128);

  k_scatter<<<dim3((NVOX + 255) / 256), dim3(256), 0, stream>>>(
      (const float*)d_in[0], (const int*)d_in[106], x5, m0, NVOX, D0, H0, W0, 5);

  const dim3 g0((M0 + 15) / 16), g1((M1 + 15) / 16), g2((M2 + 15) / 16),
             g3((M3 + 15) / 16), g4((M4 + 15) / 16);

  // conv0 + bn0 + mask + relu  (out 256x256 -> LW=LH=8, NW=1)
  k_conv_wmma<5, 3, 3, 3, 8, 8, 1, false, false><<<g0, 32, 0, stream>>>(
      x5, m0, wC0, sc0, sh0, nullptr, A16, nullptr,
      D0, H0, W0, D0, 1, 1, 1, 1, 1, 1);

  // res16 (two blocks), NW=1
  k_conv_wmma<16, 3, 3, 3, 8, 8, 1, false, false><<<g0, 32, 0, stream>>>(
      A16, m0, wR16[0], scR16[0], shR16[0], nullptr, B16, nullptr,
      D0, H0, W0, D0, 1, 1, 1, 1, 1, 1);
  k_conv_wmma<16, 3, 3, 3, 8, 8, 1, false, true><<<g0, 32, 0, stream>>>(
      B16, m0, wR16[1], scR16[1], shR16[1], A16, C16, nullptr,
      D0, H0, W0, D0, 1, 1, 1, 1, 1, 1);
  k_conv_wmma<16, 3, 3, 3, 8, 8, 1, false, false><<<g0, 32, 0, stream>>>(
      C16, m0, wR16[2], scR16[2], shR16[2], nullptr, B16, nullptr,
      D0, H0, W0, D0, 1, 1, 1, 1, 1, 1);
  k_conv_wmma<16, 3, 3, 3, 8, 8, 1, false, true><<<g0, 32, 0, stream>>>(
      B16, m0, wR16[3], scR16[3], shR16[3], C16, A16, nullptr,
      D0, H0, W0, D0, 1, 1, 1, 1, 1, 1);

  // down1: (41,256,256)x16 -> (21,128,128)x32  (LW=LH=7, NW=2)
  k_conv_wmma<16, 3, 3, 3, 7, 7, 2, true, false><<<g1, 64, 0, stream>>>(
      A16, m0, wD1, scD1, shD1, nullptr, A32, m1,
      D0, H0, W0, D1, 2, 2, 2, 1, 1, 1);

  // res32, NW=2
  k_conv_wmma<32, 3, 3, 3, 7, 7, 2, false, false><<<g1, 64, 0, stream>>>(
      A32, m1, wR32[0], scR32[0], shR32[0], nullptr, B32, nullptr,
      D1, H1, W1, D1, 1, 1, 1, 1, 1, 1);
  k_conv_wmma<32, 3, 3, 3, 7, 7, 2, false, true><<<g1, 64, 0, stream>>>(
      B32, m1, wR32[1], scR32[1], shR32[1], A32, C32, nullptr,
      D1, H1, W1, D1, 1, 1, 1, 1, 1, 1);
  k_conv_wmma<32, 3, 3, 3, 7, 7, 2, false, false><<<g1, 64, 0, stream>>>(
      C32, m1, wR32[2], scR32[2], shR32[2], nullptr, B32, nullptr,
      D1, H1, W1, D1, 1, 1, 1, 1, 1, 1);
  k_conv_wmma<32, 3, 3, 3, 7, 7, 2, false, true><<<g1, 64, 0, stream>>>(
      B32, m1, wR32[3], scR32[3], shR32[3], C32, A32, nullptr,
      D1, H1, W1, D1, 1, 1, 1, 1, 1, 1);

  // down2: (21,128,128)x32 -> (11,64,64)x64  (LW=LH=6, NW=4)
  k_conv_wmma<32, 3, 3, 3, 6, 6, 4, true, false><<<g2, 128, 0, stream>>>(
      A32, m1, wD2, scD2, shD2, nullptr, A64, m2,
      D1, H1, W1, D2, 2, 2, 2, 1, 1, 1);

  // res64, NW=4
  k_conv_wmma<64, 3, 3, 3, 6, 6, 4, false, false><<<g2, 128, 0, stream>>>(
      A64, m2, wR64[0], scR64[0], shR64[0], nullptr, B64, nullptr,
      D2, H2, W2, D2, 1, 1, 1, 1, 1, 1);
  k_conv_wmma<64, 3, 3, 3, 6, 6, 4, false, true><<<g2, 128, 0, stream>>>(
      B64, m2, wR64[1], scR64[1], shR64[1], A64, C64, nullptr,
      D2, H2, W2, D2, 1, 1, 1, 1, 1, 1);
  k_conv_wmma<64, 3, 3, 3, 6, 6, 4, false, false><<<g2, 128, 0, stream>>>(
      C64, m2, wR64[2], scR64[2], shR64[2], nullptr, B64, nullptr,
      D2, H2, W2, D2, 1, 1, 1, 1, 1, 1);
  k_conv_wmma<64, 3, 3, 3, 6, 6, 4, false, true><<<g2, 128, 0, stream>>>(
      B64, m2, wR64[3], scR64[3], shR64[3], C64, A64, nullptr,
      D2, H2, W2, D2, 1, 1, 1, 1, 1, 1);

  // down3: (11,64,64)x64 -> (5,32,32)x128, pad (0,1,1)  (LW=LH=5, NW=8)
  k_conv_wmma<64, 3, 3, 3, 5, 5, 8, true, false><<<g3, 256, 0, stream>>>(
      A64, m2, wD3, scD3, shD3, nullptr, A128, m3,
      D2, H2, W2, D3, 2, 2, 2, 0, 1, 1);

  // res128, NW=8
  k_conv_wmma<128, 3, 3, 3, 5, 5, 8, false, false><<<g3, 256, 0, stream>>>(
      A128, m3, wR128[0], scR128[0], shR128[0], nullptr, B128, nullptr,
      D3, H3, W3, D3, 1, 1, 1, 1, 1, 1);
  k_conv_wmma<128, 3, 3, 3, 5, 5, 8, false, true><<<g3, 256, 0, stream>>>(
      B128, m3, wR128[1], scR128[1], shR128[1], A128, C128, nullptr,
      D3, H3, W3, D3, 1, 1, 1, 1, 1, 1);
  k_conv_wmma<128, 3, 3, 3, 5, 5, 8, false, false><<<g3, 256, 0, stream>>>(
      C128, m3, wR128[2], scR128[2], shR128[2], nullptr, B128, nullptr,
      D3, H3, W3, D3, 1, 1, 1, 1, 1, 1);
  k_conv_wmma<128, 3, 3, 3, 5, 5, 8, false, true><<<g3, 256, 0, stream>>>(
      B128, m3, wR128[3], scR128[3], shR128[3], C128, A128, nullptr,
      D3, H3, W3, D3, 1, 1, 1, 1, 1, 1);

  // down4: kernel (3,1,1), stride (2,1,1), pad 0: (5,32,32)x128 -> (2,32,32)x128  (NW=8)
  k_conv_wmma<128, 3, 1, 1, 5, 5, 8, true, false><<<g4, 256, 0, stream>>>(
      A128, m3, wD4, scD4, shD4, nullptr, F, m4,
      D3, H3, W3, D4, 2, 1, 1, 0, 0, 0);

  // (1,2,32,32,128) -> (1,256,32,32) f32
  k_final<<<dim3((M4 * 128 + 255) / 256), dim3(256), 0, stream>>>(
      F, (float*)d_out, D4, H4, W4, 128);
}
